// T2_90134183674578
// MI455X (gfx1250) — compile-verified, matrix operand order checked by reading, not run on hardware
//
#include <hip/hip_runtime.h>

// Problem constants (from reference setup_inputs)
#define B_   16
#define C_   192
#define K2   384      // fused K = 2*C (logp2 and logp3 in one GEMM)
#define TX   384
#define TY   1536
#define NEG_INF_F (-1e9f)
#define LOG2PI_F 1.8378770664093453f

typedef __attribute__((ext_vector_type(16))) _Float16 v16h;
typedef __attribute__((ext_vector_type(8)))  float    v8f;

// ---------------------------------------------------------------------------
// Prep A-side (text axis): Aprep[b][k][x], k<C -> o_scale, k>=C -> m*o_scale
// bias[b][x] = sum_c (-0.5*log2pi - logs - 0.5*m^2*o_scale)
// ---------------------------------------------------------------------------
__global__ __launch_bounds__(TX) void prep_x_kernel(
    const float* __restrict__ m_p, const float* __restrict__ logs_p,
    _Float16* __restrict__ Aprep, float* __restrict__ bias)
{
  int b = blockIdx.x, x = threadIdx.x;
  const float* mb = m_p    + (size_t)b * C_ * TX;
  const float* lb = logs_p + (size_t)b * C_ * TX;
  _Float16*    Ab = Aprep  + (size_t)b * K2 * TX;
  float acc = 0.f;
  for (int c = 0; c < C_; ++c) {
    float ls = lb[c * TX + x];
    float mm = mb[c * TX + x];
    float os = __expf(-2.f * ls);
    Ab[c * TX + x]        = (_Float16)os;
    Ab[(C_ + c) * TX + x] = (_Float16)(mm * os);
    acc += -0.5f * LOG2PI_F - ls - 0.5f * mm * mm * os;
  }
  bias[b * TX + x] = acc;
}

// ---------------------------------------------------------------------------
// Prep B-side (mel axis): Bprep[b][k][y], k<C -> -0.5*z^2, k>=C -> z
// ---------------------------------------------------------------------------
__global__ __launch_bounds__(256) void prep_y_kernel(
    const float* __restrict__ z_p, _Float16* __restrict__ Bprep)
{
  int b = blockIdx.x;
  int y = blockIdx.y * 256 + threadIdx.x;
  const float* zb = z_p   + (size_t)b * C_ * TY;
  _Float16*    Bb = Bprep + (size_t)b * K2 * TY;
  for (int c = 0; c < C_; ++c) {
    float z = zb[c * TY + y];
    Bb[c * TY + y]        = (_Float16)(-0.5f * z * z);
    Bb[(C_ + c) * TY + y] = (_Float16)z;
  }
}

// ---------------------------------------------------------------------------
// WMMA GEMM: logp[b][y][x] = sum_k Bprep[b][k][y] * Aprep[b][k][x] + bias[b][x]
// M = Ty (rows), N = Tx (lanes) so D stores are contiguous per half-wave.
// WG = 8 waves; WG tile = 128(M) x 128(N); each wave: 16(M) x 128(N).
// LDS fragments stored [row][k] with pitch 34 halfs (conflict-free b32 reads).
// ---------------------------------------------------------------------------
#define GM_TILE 128
#define GN_TILE 128
#define KSTEP   32
#define LPITCH  34

__global__ __launch_bounds__(256) void gemm_logp_kernel(
    const _Float16* __restrict__ Aprep,   // [B][K2][TX]  (N side)
    const _Float16* __restrict__ Bprep,   // [B][K2][TY]  (M side)
    const float*    __restrict__ bias,    // [B][TX]
    float*          __restrict__ logp)    // [B][TY][TX]
{
  __shared__ _Float16 lA[GM_TILE * LPITCH];   // [m][k]  (y-local rows)
  __shared__ _Float16 lB[GN_TILE * LPITCH];   // [n][k]  (x-local rows)

  int b   = blockIdx.x;
  int m0  = blockIdx.y * GM_TILE;   // Ty offset
  int n0  = blockIdx.z * GN_TILE;   // Tx offset
  int tid = threadIdx.x;
  int wave  = tid >> 5;
  int lane  = tid & 31;
  int lhalf = lane >> 4;
  int l16   = lane & 15;

  const _Float16* gA = Bprep + (size_t)b * K2 * TY;   // M side
  const _Float16* gB = Aprep + (size_t)b * K2 * TX;   // N side

  v8f acc[8];
  for (int i = 0; i < 8; ++i) acc[i] = (v8f){0.f,0.f,0.f,0.f,0.f,0.f,0.f,0.f};

  for (int k0 = 0; k0 < K2; k0 += KSTEP) {
    __syncthreads();
    // Stage 32(K) x 128 tiles of A and B into LDS, transposed to [row][k].
    for (int i = 0; i < 16; ++i) {
      int e = tid + 256 * i;          // 0..4095
      int k = e >> 7;                 // 0..31
      int r = e & 127;                // 0..127
      lA[r * LPITCH + k] = gA[(size_t)(k0 + k) * TY + m0 + r];
      lB[r * LPITCH + k] = gB[(size_t)(k0 + k) * TX + n0 + r];
    }
    __syncthreads();

    // A fragment for this wave's 16-row M strip (documented 16-bit layout:
    // VGPR v holds K pair kb,kb+1 with kb = (v<4 ? 2v : 2v+8) + 8*half).
    v16h afrag = {};
    const _Float16* aRow = &lA[(wave * 16 + l16) * LPITCH];
    #pragma unroll
    for (int v = 0; v < 8; ++v) {
      int kb = ((v < 4) ? 2 * v : 2 * v + 8) + 8 * lhalf;
      afrag[2 * v]     = aRow[kb];
      afrag[2 * v + 1] = aRow[kb + 1];
    }

    #pragma unroll
    for (int nt = 0; nt < 8; ++nt) {
      v16h bfrag = {};
      const _Float16* bRow = &lB[(nt * 16 + l16) * LPITCH];
      #pragma unroll
      for (int v = 0; v < 8; ++v) {
        int kb = ((v < 4) ? 2 * v : 2 * v + 8) + 8 * lhalf;
        bfrag[2 * v]     = bRow[kb];
        bfrag[2 * v + 1] = bRow[kb + 1];
      }
      acc[nt] = __builtin_amdgcn_wmma_f32_16x16x32_f16(
          false, afrag, false, bfrag, (short)0, acc[nt], false, false);
    }
  }

  // Store D + bias. D layout: element r -> M = r + 8*half, N = lane&15.
  for (int nt = 0; nt < 8; ++nt) {
    int xg = n0 + nt * 16 + l16;
    float bv = bias[b * TX + xg];
    int ybase = m0 + wave * 16 + 8 * lhalf;
    float* orow = logp + (size_t)b * TY * TX;
    #pragma unroll
    for (int r = 0; r < 8; ++r) {
      orow[(size_t)(ybase + r) * TX + xg] = acc[nt][r] + bv;
    }
  }
}

// ---------------------------------------------------------------------------
// Monotonic alignment search: one wave32 per batch. Each lane keeps 12
// consecutive x-values of the DP column in registers; the x-1 shift needs a
// single __shfl_up per step (no barriers). Decision bits (vals[y-1][x] <
// vals[y-1][x-1]) are packed 12/lane into LDS for the backtrack.
// ---------------------------------------------------------------------------
__global__ __launch_bounds__(32) void mas_dp_kernel(
    const float* __restrict__ logp,    // [B][TY][TX]
    const float* __restrict__ x_mask,  // [B][1][TX]
    const float* __restrict__ y_mask,  // [B][1][TY]
    float*       __restrict__ out)     // [B][TX][TY]
{
  extern __shared__ unsigned short bits[];   // (TY+1) rows x 32 lanes
  int b = blockIdx.x;
  int lane = threadIdx.x;

  // text / mel lengths from masks
  float sx = 0.f, sy = 0.f;
  for (int j = 0; j < 12; ++j) sx += x_mask[b * TX + lane * 12 + j];
  for (int j = 0; j < 48; ++j) sy += y_mask[b * TY + lane * 48 + j];
  for (int off = 16; off; off >>= 1) {
    sx += __shfl_xor(sx, off);
    sy += __shfl_xor(sy, off);
  }
  int tx = (int)(sx + 0.5f);
  int ty = (int)(sy + 0.5f);

  float v[12];
  #pragma unroll
  for (int j = 0; j < 12; ++j) v[j] = NEG_INF_F;

  const float4* base4 = (const float4*)(logp + (size_t)b * TY * TX + lane * 12);
  const int P4 = TX / 4;   // 96 float4 per column row
  int xbase = lane * 12;

  // double-buffered column loads
  float4 c0 = base4[0], c1 = base4[1], c2 = base4[2];

  for (int y = 0; y < TY; ++y) {
    float col[12] = { c0.x, c0.y, c0.z, c0.w,
                      c1.x, c1.y, c1.z, c1.w,
                      c2.x, c2.y, c2.z, c2.w };
    if (y + 1 < TY) {
      const float4* nx = base4 + (size_t)(y + 1) * P4;
      c0 = nx[0]; c1 = nx[1]; c2 = nx[2];
    }
    if (y + 8 < TY)
      __builtin_prefetch((const void*)(base4 + (size_t)(y + 8) * P4), 0, 0);

    float left = __shfl_up(v[11], 1);
    if (lane == 0) left = (y == 0) ? 0.f : NEG_INF_F;

    float nv[12];
    #pragma unroll
    for (int j = 0; j < 12; ++j) {
      int x = xbase + j;
      float vm = (j == 0) ? left : v[j - 1];
      float vs = (x == y) ? NEG_INF_F : v[j];
      float t = fmaxf(vs, vm) + col[j];
      bool feas = (x <= y) && (x >= tx + y - ty);
      nv[j] = feas ? t : NEG_INF_F;
    }

    // decision bits for row y+1: newv[x] < newv[x-1]  (x=0 bit never used)
    float nl = __shfl_up(nv[11], 1);
    unsigned mask = 0;
    #pragma unroll
    for (int j = 0; j < 12; ++j) {
      float prev = (j == 0) ? nl : nv[j - 1];
      mask |= (nv[j] < prev) ? (1u << j) : 0u;
    }
    bits[(y + 1) * 32 + lane] = (unsigned short)mask;

    #pragma unroll
    for (int j = 0; j < 12; ++j) v[j] = nv[j];
  }

  __syncthreads();   // make LDS bits visible/ordered for lane 0

  if (lane == 0) {
    int idx = tx - 1;
    float* ob = out + (size_t)b * TX * TY;
    for (int y = TY - 1; y >= 0; --y) {
      bool active = y < ty;
      if (active) ob[(size_t)idx * TY + y] = 1.0f;
      bool dec = (idx != 0) &&
                 ((idx == y) || ((bits[y * 32 + idx / 12] >> (idx % 12)) & 1));
      if (active && dec) --idx;
    }
  }
}

// ---------------------------------------------------------------------------
extern "C" void kernel_launch(void* const* d_in, const int* in_sizes, int n_in,
                              void* d_out, int out_size, void* d_ws, size_t ws_size,
                              hipStream_t stream) {
  (void)in_sizes; (void)n_in; (void)ws_size;
  const float* z_p    = (const float*)d_in[0];
  const float* m_p    = (const float*)d_in[1];
  const float* logs_p = (const float*)d_in[2];
  const float* x_mask = (const float*)d_in[3];
  const float* y_mask = (const float*)d_in[4];

  // workspace layout (all 256B aligned by construction): total ~61.4 MB
  const size_t SZ_A    = (size_t)B_ * K2 * TX * sizeof(_Float16);  //  4,718,592
  const size_t SZ_B    = (size_t)B_ * K2 * TY * sizeof(_Float16);  // 18,874,368
  const size_t SZ_BIAS = (size_t)B_ * TX * sizeof(float);          //     24,576
  char* ws = (char*)d_ws;
  _Float16* Aprep = (_Float16*)(ws);
  _Float16* Bprep = (_Float16*)(ws + SZ_A);
  float*    bias  = (float*)   (ws + SZ_A + SZ_B);
  float*    logp  = (float*)   (ws + SZ_A + SZ_B + SZ_BIAS);

  prep_x_kernel<<<B_, TX, 0, stream>>>(m_p, logs_p, Aprep, bias);
  prep_y_kernel<<<dim3(B_, TY / 256), 256, 0, stream>>>(z_p, Bprep);
  gemm_logp_kernel<<<dim3(B_, TY / GM_TILE, TX / GN_TILE), 256, 0, stream>>>(
      Aprep, Bprep, bias, logp);

  // clear the hard-alignment output (mostly zeros; backtrack scatters 1s)
  hipMemsetAsync(d_out, 0, (size_t)out_size * sizeof(float), stream);

  const int dpLds = (TY + 1) * 32 * (int)sizeof(unsigned short);   // 98,368 B
  hipFuncSetAttribute((const void*)mas_dp_kernel,
                      hipFuncAttributeMaxDynamicSharedMemorySize, dpLds);
  mas_dp_kernel<<<B_, 32, dpLds, stream>>>(logp, x_mask, y_mask, (float*)d_out);
}
